// CryptoGNN_46136538693914
// MI455X (gfx1250) — matrix operand order repaired and checked by reading, not compiled, not run on hardware
//
#include <hip/hip_runtime.h>
#include <hip/hip_bf16.h>

#define N_NODES 100000
#define N_EDGES 600000
#define BATCH   64
#define IN_D    6
#define H_D     128
#define EMB_D   64
#define KD_D    16
#define CD_D    8
#define LD_D    6
#define OUT_D   3
#define S_D     16

typedef __attribute__((ext_vector_type(16))) _Float16 v16h;
typedef __attribute__((ext_vector_type(8)))  float    v8f;

// ---------------------------------------------------------------- utilities
__global__ void fill_f32(float* p, float v, size_t n) {
  size_t i = (size_t)blockIdx.x * blockDim.x + threadIdx.x;
  if (i < n) p[i] = v;
}

// ---------------------------------------------------------------- degree / norm
__global__ void deg_accum(const int* __restrict__ dst, float* __restrict__ deg) {
  int e = blockIdx.x * blockDim.x + threadIdx.x;
  if (e < N_EDGES) atomicAdd(&deg[dst[e]], 1.0f);
}

__global__ void inv_sqrt_deg(const float* __restrict__ deg, float* __restrict__ dis) {
  int i = blockIdx.x * blockDim.x + threadIdx.x;
  if (i < N_NODES) dis[i] = rsqrtf(fmaxf(deg[i], 1.0f));
}

// ---------------------------------------------------------------- m = x @ W1  (K=6)
__global__ void proj_in(const float* __restrict__ x, const float* __restrict__ W1,
                        float* __restrict__ m) {
  size_t gid = (size_t)blockIdx.x * blockDim.x + threadIdx.x;
  if (gid >= (size_t)N_NODES * H_D) return;
  size_t i = gid / H_D;
  int    f = (int)(gid % H_D);
  float acc = 0.0f;
#pragma unroll
  for (int t = 0; t < IN_D; ++t)
    acc += x[i * IN_D + t] * W1[t * H_D + f];
  m[gid] = acc;
}

// ---------------------------------------------------------------- edge scatter
// One wave32 per edge; lane covers 4 features (float4 gather + 4 f32 atomics).
// Edges [E, E+N) are self-loops.
__global__ __launch_bounds__(256)
void scatter_edges(const int* __restrict__ src, const int* __restrict__ dst,
                   const float* __restrict__ dis,
                   const float* __restrict__ m, float* __restrict__ agg) {
  int e = blockIdx.x * 8 + (threadIdx.x >> 5);
  if (e >= N_EDGES + N_NODES) return;
  int s, d;
  if (e < N_EDGES) { s = src[e]; d = dst[e]; }
  else             { s = e - N_EDGES; d = s; }
  float nr = dis[s] * dis[d];
  int f = (threadIdx.x & 31) * 4;
  const float4 v = *(const float4*)(m + (size_t)s * H_D + f);
  float* ap = agg + (size_t)d * H_D + f;
  atomicAdd(ap + 0, v.x * nr);
  atomicAdd(ap + 1, v.y * nr);
  atomicAdd(ap + 2, v.z * nr);
  atomicAdd(ap + 3, v.w * nr);
}

// ---------------------------------------------------------------- activations
// h1 = relu(agg + b1) in f32 (for residual) and f16 (for WMMA GEMM)
__global__ void bias_relu_h(const float* __restrict__ agg, const float* __restrict__ b,
                            float* __restrict__ h, _Float16* __restrict__ hh) {
  size_t gid = (size_t)blockIdx.x * blockDim.x + threadIdx.x;
  if (gid >= (size_t)N_NODES * H_D) return;
  int f = (int)(gid % H_D);
  float r = fmaxf(agg[gid] + b[f], 0.0f);
  h[gid]  = r;
  hh[gid] = (_Float16)r;
}

__global__ void bias_residual(const float* __restrict__ agg, const float* __restrict__ b,
                              const float* __restrict__ h1, float* __restrict__ h2) {
  size_t gid = (size_t)blockIdx.x * blockDim.x + threadIdx.x;
  if (gid >= (size_t)N_NODES * H_D) return;
  int f = (int)(gid % H_D);
  h2[gid] = agg[gid] + b[f] + h1[gid];
}

// ---------------------------------------------------------------- W2 -> BT f16 [n][k]
__global__ void w2_transpose_cvt(const float* __restrict__ W2, _Float16* __restrict__ BT) {
  int gid = blockIdx.x * blockDim.x + threadIdx.x;
  if (gid >= H_D * H_D) return;
  int n = gid / H_D, k = gid % H_D;
  BT[n * H_D + k] = (_Float16)W2[k * H_D + n];
}

// ---------------------------------------------------------------- WMMA GEMM
// C[N,128] = Ah[N,128] (f16) @ W2 (as BT f16 [n][k]).
// Block = 256 threads (8 waves) covers a 32-row x 128-col region.
// BT (32KB) + 32-row A stripe (8KB) staged in LDS; each wave owns one
// 16-col tile and computes two 16x16 tiles, 4 WMMAs each over K=128.
__global__ __launch_bounds__(256)
void gemm_h_wmma(const _Float16* __restrict__ Ah, const _Float16* __restrict__ BT,
                 float* __restrict__ C) {
  __shared__ _Float16 sB[H_D * H_D];   // [n][k], 32 KB
  __shared__ _Float16 sA[32 * H_D];    // [row][k], 8 KB

  const int tid = threadIdx.x;
  const size_t row0 = (size_t)blockIdx.x * 32;

  // stage BT: 16384 halves = 2048 uint4
  {
    const uint4* gB = (const uint4*)BT;
    uint4* lB = (uint4*)sB;
#pragma unroll
    for (int i = 0; i < 8; ++i) lB[tid + i * 256] = gB[tid + i * 256];
  }
  // stage A stripe: 4096 halves = 512 uint4
  {
    const uint4* gA = (const uint4*)(Ah + row0 * H_D);
    uint4* lA = (uint4*)sA;
#pragma unroll
    for (int i = 0; i < 2; ++i) lA[tid + i * 256] = gA[tid + i * 256];
  }
  __syncthreads();

  const int wave = tid >> 5;           // 0..7 -> column tile
  const int lane = tid & 31;           // all lanes active around WMMA
  const int half = lane >> 4;          // 0/1
  const int l15  = lane & 15;
  const int n    = wave * 16 + l15;

  const _Float16* brow = sB + (size_t)n * H_D;

#pragma unroll
  for (int rt = 0; rt < 2; ++rt) {
    const _Float16* arow = sA + (size_t)(rt * 16 + l15) * H_D;
    v8f acc = {};
#pragma unroll
    for (int k0 = 0; k0 < H_D; k0 += 32) {
      // A fragment (16-bit A 16x32 layout): lane group `half` holds
      // K = k0+half*8..+7 in v0..3 and K = k0+16+half*8..+7 in v4..7.
      union { v16h v; uint4 q[2]; } ua;
      ua.q[0] = *(const uint4*)(arow + k0 + half * 8);
      ua.q[1] = *(const uint4*)(arow + k0 + 16 + half * 8);
      // B fragment: lane = column n, same K order as A (k0+half*8..+7, k0+16+half*8..+7).
      union { v16h v; uint4 q[2]; } ub;
      ub.q[0] = *(const uint4*)(brow + k0 + half * 8);
      ub.q[1] = *(const uint4*)(brow + k0 + 16 + half * 8);

      acc = __builtin_amdgcn_wmma_f32_16x16x32_f16(
          /*neg_a=*/false, ua.v, /*neg_b=*/false, ub.v,
          /*c_mod=*/(short)0, acc, /*reuse_a=*/false, /*reuse_b=*/false);
    }
    // C/D layout: lanes 0-15 -> M = base+r, lanes 16-31 -> M = base+8+r, N = l15.
    const size_t mbase = row0 + rt * 16 + (size_t)half * 8;
#pragma unroll
    for (int r = 0; r < 8; ++r)
      C[(mbase + r) * H_D + n] = acc[r];
  }
}

// ---------------------------------------------------------------- pooling
__global__ void pool_sum(const float* __restrict__ h2, const int* __restrict__ bidx,
                         float* __restrict__ g_sum) {
  size_t gid = (size_t)blockIdx.x * blockDim.x + threadIdx.x;
  if (gid >= (size_t)N_NODES * H_D) return;
  size_t i = gid / H_D;
  int    f = (int)(gid % H_D);
  atomicAdd(&g_sum[(size_t)bidx[i] * H_D + f], h2[gid]);
}

__global__ void pool_cnt(const int* __restrict__ bidx, float* __restrict__ cnt) {
  int i = blockIdx.x * blockDim.x + threadIdx.x;
  if (i < N_NODES) atomicAdd(&cnt[bidx[i]], 1.0f);
}

// ---------------------------------------------------------------- head kernels
__global__ void graph_proj(const float* __restrict__ g_sum, const float* __restrict__ cnt,
                           const float* __restrict__ Wg, const float* __restrict__ bg,
                           float* __restrict__ g) {
  int gid = blockIdx.x * blockDim.x + threadIdx.x;
  if (gid >= BATCH * H_D) return;
  int b = gid / H_D, j = gid % H_D;
  float inv = 1.0f / fmaxf(cnt[b], 1.0f);
  float acc = bg[j];
  for (int t = 0; t < H_D; ++t)
    acc += g_sum[(size_t)b * H_D + t] * inv * Wg[t * H_D + j];
  g[gid] = acc;
}

__global__ void seq_mean(const int* __restrict__ ti, const int* __restrict__ ki,
                         const int* __restrict__ vi,
                         const float* __restrict__ Et, const float* __restrict__ Ek,
                         const float* __restrict__ Ev, float* __restrict__ sm) {
  int gid = blockIdx.x * blockDim.x + threadIdx.x;
  if (gid >= BATCH * 3 * EMB_D) return;
  int b = gid / (3 * EMB_D), t = gid % (3 * EMB_D);
  float acc = 0.0f;
  if (t < EMB_D) {
    for (int s = 0; s < S_D; ++s) acc += Et[ti[b * S_D + s] * EMB_D + t];
  } else if (t < 2 * EMB_D) {
    for (int s = 0; s < S_D; ++s) acc += Ek[ki[b * S_D + s] * EMB_D + (t - EMB_D)];
  } else {
    for (int s = 0; s < S_D; ++s) acc += Ev[vi[b * S_D + s] * EMB_D + (t - 2 * EMB_D)];
  }
  sm[gid] = acc * (1.0f / (float)S_D);
}

__global__ void pragma_proj(const float* __restrict__ sm, const float* __restrict__ Wp,
                            const float* __restrict__ bp, float* __restrict__ p) {
  int gid = blockIdx.x * blockDim.x + threadIdx.x;
  if (gid >= BATCH * H_D) return;
  int b = gid / H_D, j = gid % H_D;
  float acc = bp[j];
  for (int t = 0; t < 3 * EMB_D; ++t)
    acc += sm[b * 3 * EMB_D + t] * Wp[t * H_D + j];
  p[gid] = fmaxf(acc, 0.0f);
}

__global__ void kcl_proj(const int* __restrict__ kid, const float* __restrict__ cond,
                         const float* __restrict__ loc,
                         const float* __restrict__ Ekid,
                         const float* __restrict__ Wc, const float* __restrict__ bc,
                         const float* __restrict__ Wl, const float* __restrict__ bl,
                         float* __restrict__ kv, float* __restrict__ cv,
                         float* __restrict__ lv) {
  int gid = blockIdx.x * blockDim.x + threadIdx.x;
  if (gid >= BATCH * KD_D) return;
  int b = gid / KD_D, j = gid % KD_D;
  kv[gid] = Ekid[kid[b] * KD_D + j];
  float ac = bc[j];
#pragma unroll
  for (int t = 0; t < CD_D; ++t) ac += cond[b * CD_D + t] * Wc[t * KD_D + j];
  cv[gid] = fmaxf(ac, 0.0f);
  float al = bl[j];
#pragma unroll
  for (int t = 0; t < LD_D; ++t) al += loc[b * LD_D + t] * Wl[t * KD_D + j];
  lv[gid] = fmaxf(al, 0.0f);
}

__global__ void fuse1(const float* __restrict__ g, const float* __restrict__ p,
                      const float* __restrict__ kv, const float* __restrict__ cv,
                      const float* __restrict__ lv,
                      const float* __restrict__ Wm1, const float* __restrict__ bm1,
                      float* __restrict__ hid) {
  int gid = blockIdx.x * blockDim.x + threadIdx.x;
  if (gid >= BATCH * H_D) return;
  int b = gid / H_D, j = gid % H_D;
  float acc = bm1[j];
  for (int t = 0; t < H_D; ++t)  acc += g[b * H_D + t]  * Wm1[t * H_D + j];
  for (int t = 0; t < H_D; ++t)  acc += p[b * H_D + t]  * Wm1[(H_D + t) * H_D + j];
#pragma unroll
  for (int t = 0; t < KD_D; ++t) acc += kv[b * KD_D + t] * Wm1[(2 * H_D + t) * H_D + j];
#pragma unroll
  for (int t = 0; t < KD_D; ++t) acc += cv[b * KD_D + t] * Wm1[(2 * H_D + KD_D + t) * H_D + j];
#pragma unroll
  for (int t = 0; t < KD_D; ++t) acc += lv[b * KD_D + t] * Wm1[(2 * H_D + 2 * KD_D + t) * H_D + j];
  hid[gid] = fmaxf(acc, 0.0f);
}

__global__ void fuse2(const float* __restrict__ hid, const float* __restrict__ Wm2,
                      const float* __restrict__ bm2, float* __restrict__ out) {
  int gid = blockIdx.x * blockDim.x + threadIdx.x;
  if (gid >= BATCH * OUT_D) return;
  int b = gid / OUT_D, o = gid % OUT_D;
  float acc = bm2[o];
  for (int t = 0; t < H_D; ++t) acc += hid[b * H_D + t] * Wm2[t * OUT_D + o];
  out[gid] = acc;
}

// ---------------------------------------------------------------- launch
extern "C" void kernel_launch(void* const* d_in, const int* in_sizes, int n_in,
                              void* d_out, int out_size, void* d_ws, size_t ws_size,
                              hipStream_t stream) {
  (void)in_sizes; (void)n_in; (void)out_size; (void)ws_size;

  const float* x        = (const float*)d_in[0];
  const int*   ei       = (const int*)d_in[1];
  const int*   src      = ei;
  const int*   dst      = ei + N_EDGES;
  const int*   bidx     = (const int*)d_in[2];
  const int*   ti       = (const int*)d_in[3];
  const int*   ki       = (const int*)d_in[4];
  const int*   vi       = (const int*)d_in[5];
  const int*   kid      = (const int*)d_in[6];
  const float* cond     = (const float*)d_in[7];
  const float* loc      = (const float*)d_in[8];
  const float* W1       = (const float*)d_in[9];
  const float* b1       = (const float*)d_in[10];
  const float* W2       = (const float*)d_in[11];
  const float* b2       = (const float*)d_in[12];
  const float* Wg       = (const float*)d_in[13];
  const float* bg       = (const float*)d_in[14];
  const float* Et       = (const float*)d_in[15];
  const float* Ek       = (const float*)d_in[16];
  const float* Ev       = (const float*)d_in[17];
  const float* Wp       = (const float*)d_in[18];
  const float* bp       = (const float*)d_in[19];
  const float* Ekid     = (const float*)d_in[20];
  const float* Wc       = (const float*)d_in[21];
  const float* bc       = (const float*)d_in[22];
  const float* Wl       = (const float*)d_in[23];
  const float* bl       = (const float*)d_in[24];
  const float* Wm1      = (const float*)d_in[25];
  const float* bm1      = (const float*)d_in[26];
  const float* Wm2      = (const float*)d_in[27];
  const float* bm2      = (const float*)d_in[28];
  float* out = (float*)d_out;

  // scratch carve-out
  char* ws = (char*)d_ws;
  size_t off = 0;
  auto carve = [&](size_t bytes) -> void* {
    void* p = ws + off;
    off += (bytes + 255) & ~(size_t)255;
    return p;
  };
  float*    deg    = (float*)carve((size_t)N_NODES * 4);
  float*    dis    = (float*)carve((size_t)N_NODES * 4);
  float*    m      = (float*)carve((size_t)N_NODES * H_D * 4);  // m1 -> m2 -> h2
  float*    agg    = (float*)carve((size_t)N_NODES * H_D * 4);
  float*    h1     = (float*)carve((size_t)N_NODES * H_D * 4);
  _Float16* h1h    = (_Float16*)carve((size_t)N_NODES * H_D * 2);
  _Float16* BT     = (_Float16*)carve((size_t)H_D * H_D * 2);
  float*    g_sum  = (float*)carve((size_t)BATCH * H_D * 4);
  float*    cnt    = (float*)carve((size_t)BATCH * 4);
  float*    g      = (float*)carve((size_t)BATCH * H_D * 4);
  float*    sm     = (float*)carve((size_t)BATCH * 3 * EMB_D * 4);
  float*    p      = (float*)carve((size_t)BATCH * H_D * 4);
  float*    kv     = (float*)carve((size_t)BATCH * KD_D * 4);
  float*    cv     = (float*)carve((size_t)BATCH * KD_D * 4);
  float*    lv     = (float*)carve((size_t)BATCH * KD_D * 4);
  float*    hid    = (float*)carve((size_t)BATCH * H_D * 4);

  const size_t NH = (size_t)N_NODES * H_D;
  const int TB = 256;
  const int gNH = (int)((NH + TB - 1) / TB);
  const int gN  = (N_NODES + TB - 1) / TB;
  const int gE  = (N_EDGES + TB - 1) / TB;
  const int gSc = (N_EDGES + N_NODES + 7) / 8;   // 8 edges (waves) per block

  // degree + symmetric norm
  fill_f32<<<gN, TB, 0, stream>>>(deg, 1.0f, (size_t)N_NODES);   // self-loop
  deg_accum<<<gE, TB, 0, stream>>>(dst, deg);
  inv_sqrt_deg<<<gN, TB, 0, stream>>>(deg, dis);

  // layer 1: project, scatter, relu (+ f16 copy for WMMA)
  proj_in<<<gNH, TB, 0, stream>>>(x, W1, m);
  fill_f32<<<gNH, TB, 0, stream>>>(agg, 0.0f, NH);
  scatter_edges<<<gSc, TB, 0, stream>>>(src, dst, dis, m, agg);
  bias_relu_h<<<gNH, TB, 0, stream>>>(agg, b1, h1, h1h);

  // layer 2: WMMA GEMM (LDS-staged), scatter, residual
  w2_transpose_cvt<<<(H_D * H_D + TB - 1) / TB, TB, 0, stream>>>(W2, BT);
  gemm_h_wmma<<<N_NODES / 32, TB, 0, stream>>>(h1h, BT, m);      // 3125 blocks
  fill_f32<<<gNH, TB, 0, stream>>>(agg, 0.0f, NH);
  scatter_edges<<<gSc, TB, 0, stream>>>(src, dst, dis, m, agg);
  bias_residual<<<gNH, TB, 0, stream>>>(agg, b2, h1, m);         // m now holds h2

  // pooling
  fill_f32<<<(BATCH * H_D + TB - 1) / TB, TB, 0, stream>>>(g_sum, 0.0f, (size_t)BATCH * H_D);
  fill_f32<<<1, TB, 0, stream>>>(cnt, 0.0f, (size_t)BATCH);
  pool_cnt<<<gN, TB, 0, stream>>>(bidx, cnt);
  pool_sum<<<gNH, TB, 0, stream>>>(m, bidx, g_sum);
  graph_proj<<<(BATCH * H_D + TB - 1) / TB, TB, 0, stream>>>(g_sum, cnt, Wg, bg, g);

  // pragma encoder + small embeddings
  seq_mean<<<(BATCH * 3 * EMB_D + TB - 1) / TB, TB, 0, stream>>>(ti, ki, vi, Et, Ek, Ev, sm);
  pragma_proj<<<(BATCH * H_D + TB - 1) / TB, TB, 0, stream>>>(sm, Wp, bp, p);
  kcl_proj<<<(BATCH * KD_D + TB - 1) / TB, TB, 0, stream>>>(kid, cond, loc, Ekid,
                                                            Wc, bc, Wl, bl, kv, cv, lv);

  // fusion head
  fuse1<<<(BATCH * H_D + TB - 1) / TB, TB, 0, stream>>>(g, p, kv, cv, lv, Wm1, bm1, hid);
  fuse2<<<(BATCH * OUT_D + TB - 1) / TB, TB, 0, stream>>>(hid, Wm2, bm2, out);
}
